// network_18365280158192
// MI455X (gfx1250) — compile-verified, matrix operand order checked by reading, not run on hardware
//
#include <hip/hip_runtime.h>

// ---------------- model dims ----------------
#define T_STEPS 500
#define BATCH   128
#define H       400
#define HP      416      // H padded to mult of 32
#define KIN2    464      // H + A
#define KIN2P   480
#define KOUT    1264     // 3H + A
#define KOUTP   1280
#define GATES   1200     // 3H
#define A_SZ    64
#define KMIX    10
#define OUTD    121
#define U_LEN   50
#define INPD    3
#define IH1COLS 67       // INP + A

typedef __attribute__((ext_vector_type(16))) __bf16 v16bf;
typedef __attribute__((ext_vector_type(8)))  float  v8f;

__device__ inline unsigned short f2bf(float f) {
    unsigned u = __float_as_uint(f);
    unsigned r = u + 0x7fffu + ((u >> 16) & 1u);   // round-to-nearest-even
    return (unsigned short)(r >> 16);
}

struct Params {
    // inputs (fp32 unless noted)
    const float* seq;        // [T,B,3]
    const int*   char_ids;   // [B,U]
    const float *Wih1, *bih1, *bhh1, *Wc, *bc;
    const float *bih2, *bhh2, *bih3, *bhh3, *bl;
    // fp32 state / gate scratch
    float *h1, *h2, *h3, *kappa, *gi, *gh;
    // bf16 (ushort) activations, zero-padded
    unsigned short *h1b, *h2b, *h3b;       // [128, HP]
    unsigned short *catA2, *catA3;         // [128, KIN2P] relu(cat(h,w))
    unsigned short *catL;                  // [128, KOUTP] relu(cat(h3,h2,h1,w))
    // bf16 weights, zero-padded rows of length Kp
    unsigned short *Whh1b, *Whh2b, *Whh3b; // [1200, HP]
    unsigned short *Wih2b, *Wih3b;         // [1200, KIN2P]
    unsigned short *Wlb;                   // [128,  KOUTP]
    float* out;                            // [B*T, OUTD]
    unsigned *counter, *gen;
    int nwg;
};

// ---------------- grid-wide barrier ----------------
__device__ inline void gsync(const Params& P) {
    __syncthreads();
    if (threadIdx.x == 0) {
        __threadfence();
        unsigned g = __hip_atomic_load(P.gen, __ATOMIC_ACQUIRE, __HIP_MEMORY_SCOPE_AGENT);
        if (atomicAdd(P.counter, 1u) == (unsigned)(P.nwg - 1)) {
            *P.counter = 0u;
            __threadfence();
            atomicAdd(P.gen, 1u);
        } else {
            while (__hip_atomic_load(P.gen, __ATOMIC_ACQUIRE, __HIP_MEMORY_SCOPE_AGENT) == g) {
                __builtin_amdgcn_s_sleep(2);
            }
        }
        __threadfence();
    }
    __syncthreads();
}

// ------- bf16 WMMA GEMM, 2x1 register-blocked:  C[M,N] = A[M,K] * B[N,K]^T -------
// A row-major [*, lda] bf16 ; B row-major [N, ldb] bf16 (weights W) ; C fp32 [*, ldc]
// Each task = two M-tiles (32 rows) x one N-tile (16 cols); B fragment shared.
__device__ inline void gemm_bf16_m2(const unsigned short* __restrict__ A, int lda,
                                    const unsigned short* __restrict__ B, int ldb,
                                    float* __restrict__ C, int ldc,
                                    int mt2, int nt, int kIters,
                                    int gwave, int nwaves, int lane) {
    const int ntasks = mt2 * nt;
    const int lrow = lane & 15;
    const int lhi  = lane >> 4;          // 0 or 1
    for (int t = gwave; t < ntasks; t += nwaves) {
        const int m0 = (t / nt) * 32;
        const int n0 = (t % nt) * 16;
        v8f acc0 = {}, acc1 = {};
        const unsigned short* arow0 = A + (size_t)(m0 + lrow) * lda + lhi * 8;
        const unsigned short* arow1 = arow0 + (size_t)16 * lda;
        const unsigned short* brow  = B + (size_t)(n0 + lrow) * ldb + lhi * 8;
        for (int k = 0; k < kIters; ++k) {
            __builtin_prefetch((const void*)(brow + 128), 0, 1);
            union { uint4 u[2]; v16bf v; } a0, a1, b;
            a0.u[0] = *(const uint4*)(arow0);
            a0.u[1] = *(const uint4*)(arow0 + 16);
            a1.u[0] = *(const uint4*)(arow1);
            a1.u[1] = *(const uint4*)(arow1 + 16);
            b.u[0]  = *(const uint4*)(brow);
            b.u[1]  = *(const uint4*)(brow + 16);
            acc0 = __builtin_amdgcn_wmma_f32_16x16x32_bf16(
                false, a0.v, false, b.v, (short)0, acc0, false, false);
            acc1 = __builtin_amdgcn_wmma_f32_16x16x32_bf16(
                false, a1.v, false, b.v, (short)0, acc1, false, false);
            arow0 += 32; arow1 += 32; brow += 32;
        }
        float* crow0 = C + (size_t)(m0 + lhi * 8) * ldc + n0 + lrow;
        float* crow1 = crow0 + (size_t)16 * ldc;
#pragma unroll
        for (int i = 0; i < 8; ++i) crow0[(size_t)i * ldc] = acc0[i];
#pragma unroll
        for (int i = 0; i < 8; ++i) crow1[(size_t)i * ldc] = acc1[i];
    }
}

// ---------------- GRU gate update ----------------
__device__ inline void gru_update(const Params& P,
                                  const float* bih, const float* bhh,
                                  float* h, unsigned short* hb,
                                  unsigned short* relu1, int ld1, int off1,
                                  unsigned short* relu2, int ld2, int off2,
                                  int tid, int nthreads) {
    for (int idx = tid; idx < BATCH * H; idx += nthreads) {
        const int b = idx / H, j = idx - b * H;
        const size_t g0 = (size_t)b * GATES + j;
        float ir = P.gi[g0]         + bih[j];
        float iz = P.gi[g0 + H]     + bih[j + H];
        float in = P.gi[g0 + 2 * H] + bih[j + 2 * H];
        float hr = P.gh[g0]         + bhh[j];
        float hz = P.gh[g0 + H]     + bhh[j + H];
        float hn = P.gh[g0 + 2 * H] + bhh[j + 2 * H];
        float r = 1.f / (1.f + __expf(-(ir + hr)));
        float z = 1.f / (1.f + __expf(-(iz + hz)));
        float n = tanhf(in + r * hn);
        float hv = (1.f - z) * n + z * h[idx];
        h[idx] = hv;
        hb[(size_t)b * HP + j] = f2bf(hv);
        unsigned short rb = f2bf(fmaxf(hv, 0.f));
        if (relu1) relu1[(size_t)b * ld1 + off1 + j] = rb;
        if (relu2) relu2[(size_t)b * ld2 + off2 + j] = rb;
    }
}

// ---------------- persistent RNN kernel ----------------
__global__ __launch_bounds__(256) void rnn_persistent(Params P) {
    const int tid      = blockIdx.x * blockDim.x + threadIdx.x;
    const int nthreads = gridDim.x * blockDim.x;
    const int lane     = threadIdx.x & 31;
    const int gwave    = tid >> 5;
    const int nwaves   = nthreads >> 5;
    __shared__ float lds_w[8][A_SZ];

    for (int t = 0; t < T_STEPS; ++t) {
        // ---- Phase A: gi1 = x_t @ Wih1[:, :3]^T ; gh1 = h1 @ Whh1^T ----
        {
            const float* xbase = P.seq + (size_t)t * BATCH * INPD;
            for (int idx = tid; idx < BATCH * GATES; idx += nthreads) {
                const int b = idx / GATES, j = idx - b * GATES;
                const float* x = xbase + b * INPD;
                const float* w = P.Wih1 + (size_t)j * IH1COLS;
                P.gi[idx] = x[0] * w[0] + x[1] * w[1] + x[2] * w[2];
            }
            gemm_bf16_m2(P.h1b, HP, P.Whh1b, HP, P.gh, GATES, 4, 75, HP / 32,
                         gwave, nwaves, lane);
        }
        gsync(P);

        // ---- Phase B: h1 update ----
        gru_update(P, P.bih1, P.bhh1, P.h1, P.h1b,
                   P.catA2, KIN2P, 0, P.catL, KOUTP, 800, tid, nthreads);
        gsync(P);

        // ---- Phase C: Gaussian attention window (one wave per batch row) ----
        if (gwave < BATCH) {
            const int b = gwave;
            float p = 0.f;
            if (lane < 3 * KMIX) {
                const float* wc  = P.Wc + (size_t)lane * H;
                const float* h1r = P.h1 + (size_t)b * H;
                float acc = 0.f;
                for (int k = 0; k < H; ++k) acc += h1r[k] * wc[k];
                p = acc + P.bc[lane];
            }
            float p20 = __shfl(p, lane + 20);
            float kap = 0.f;
            if (lane < KMIX) {
                kap = P.kappa[(size_t)b * KMIX + lane] + __expf(p20);
                P.kappa[(size_t)b * KMIX + lane] = kap;
            }
            float phi0 = 0.f, phi1 = 0.f;
            for (int k = 0; k < KMIX; ++k) {
                float al = __expf(__shfl(p, k));
                float be = __expf(__shfl(p, 10 + k));
                float kk = __shfl(kap, k);
                float d0 = kk - (float)lane;
                float d1 = kk - (float)(lane + 32);
                phi0 += al * __expf(-be * d0 * d0);
                phi1 += al * __expf(-be * d1 * d1);
            }
            const int wslot = threadIdx.x >> 5;
            lds_w[wslot][lane] = 0.f;
            lds_w[wslot][lane + 32] = 0.f;
            __threadfence_block();
            atomicAdd(&lds_w[wslot][P.char_ids[b * U_LEN + lane]], phi0);
            if (lane < U_LEN - 32)
                atomicAdd(&lds_w[wslot][P.char_ids[b * U_LEN + lane + 32]], phi1);
            __threadfence_block();
            float w0 = fmaxf(lds_w[wslot][lane], 0.f);
            float w1 = fmaxf(lds_w[wslot][lane + 32], 0.f);
            unsigned short b0 = f2bf(w0), b1 = f2bf(w1);
            P.catA2[(size_t)b * KIN2P + H + lane] = b0;
            P.catA2[(size_t)b * KIN2P + H + lane + 32] = b1;
            P.catA3[(size_t)b * KIN2P + H + lane] = b0;
            P.catA3[(size_t)b * KIN2P + H + lane + 32] = b1;
            P.catL[(size_t)b * KOUTP + 1200 + lane] = b0;
            P.catL[(size_t)b * KOUTP + 1200 + lane + 32] = b1;
        }
        gsync(P);

        // ---- Phase D: gi2 = relu(cat(h1,w)) @ Wih2^T ; gh2 = h2 @ Whh2^T ----
        gemm_bf16_m2(P.catA2, KIN2P, P.Wih2b, KIN2P, P.gi, GATES, 4, 75, KIN2P / 32,
                     gwave, nwaves, lane);
        gemm_bf16_m2(P.h2b, HP, P.Whh2b, HP, P.gh, GATES, 4, 75, HP / 32,
                     gwave, nwaves, lane);
        gsync(P);

        // ---- Phase E: h2 update ----
        gru_update(P, P.bih2, P.bhh2, P.h2, P.h2b,
                   P.catA3, KIN2P, 0, P.catL, KOUTP, 400, tid, nthreads);
        gsync(P);

        // ---- Phase F: gi3 / gh3 ----
        gemm_bf16_m2(P.catA3, KIN2P, P.Wih3b, KIN2P, P.gi, GATES, 4, 75, KIN2P / 32,
                     gwave, nwaves, lane);
        gemm_bf16_m2(P.h3b, HP, P.Whh3b, HP, P.gh, GATES, 4, 75, HP / 32,
                     gwave, nwaves, lane);
        gsync(P);

        // ---- Phase G: h3 update ----
        gru_update(P, P.bih3, P.bhh3, P.h3, P.h3b,
                   P.catL, KOUTP, 0, nullptr, 0, 0, tid, nthreads);
        gsync(P);

        // ---- Phase H: out = relu(cat(h3,h2,h1,w)) @ Wl^T + bl (2x1 blocked) ----
        {
            const int lrow = lane & 15, lhi = lane >> 4;
            for (int tt = gwave; tt < 32; tt += nwaves) {
                const int m0 = (tt >> 3) * 32;
                const int n0 = (tt & 7) * 16;
                v8f acc0 = {}, acc1 = {};
                const unsigned short* arow0 =
                    P.catL + (size_t)(m0 + lrow) * KOUTP + lhi * 8;
                const unsigned short* arow1 = arow0 + (size_t)16 * KOUTP;
                const unsigned short* brow =
                    P.Wlb + (size_t)(n0 + lrow) * KOUTP + lhi * 8;
                for (int k = 0; k < KOUTP / 32; ++k) {
                    union { uint4 u[2]; v16bf v; } a0, a1, b;
                    a0.u[0] = *(const uint4*)(arow0);
                    a0.u[1] = *(const uint4*)(arow0 + 16);
                    a1.u[0] = *(const uint4*)(arow1);
                    a1.u[1] = *(const uint4*)(arow1 + 16);
                    b.u[0]  = *(const uint4*)(brow);
                    b.u[1]  = *(const uint4*)(brow + 16);
                    acc0 = __builtin_amdgcn_wmma_f32_16x16x32_bf16(
                        false, a0.v, false, b.v, (short)0, acc0, false, false);
                    acc1 = __builtin_amdgcn_wmma_f32_16x16x32_bf16(
                        false, a1.v, false, b.v, (short)0, acc1, false, false);
                    arow0 += 32; arow1 += 32; brow += 32;
                }
                const int o = n0 + lrow;
                if (o < OUTD) {
                    const float bias = P.bl[o];
#pragma unroll
                    for (int i = 0; i < 8; ++i) {
                        const int m = m0 + lhi * 8 + i;
                        P.out[((size_t)m * T_STEPS + t) * OUTD + o] = acc0[i] + bias;
                        P.out[((size_t)(m + 16) * T_STEPS + t) * OUTD + o] = acc1[i] + bias;
                    }
                }
            }
        }
        gsync(P);
    }
}

// ---------------- fp32 -> bf16 weight convert with zero padding ----------------
__global__ void convert_pad(const float* __restrict__ src, unsigned short* __restrict__ dst,
                            int rows, int srcRows, int K, int Kp) {
    const int total = rows * Kp;
    for (int idx = blockIdx.x * blockDim.x + threadIdx.x; idx < total;
         idx += gridDim.x * blockDim.x) {
        const int r = idx / Kp, k = idx - r * Kp;
        float v = (r < srcRows && k < K) ? src[(size_t)r * K + k] : 0.f;
        dst[idx] = f2bf(v);
    }
}

extern "C" void kernel_launch(void* const* d_in, const int* in_sizes, int n_in,
                              void* d_out, int out_size, void* d_ws, size_t ws_size,
                              hipStream_t stream) {
    (void)in_sizes; (void)n_in; (void)out_size; (void)ws_size;
    Params P;
    P.seq      = (const float*)d_in[0];
    P.char_ids = (const int*)  d_in[1];
    P.Wih1 = (const float*)d_in[2];
    const float* Whh1 = (const float*)d_in[3];
    P.bih1 = (const float*)d_in[4];
    P.bhh1 = (const float*)d_in[5];
    P.Wc   = (const float*)d_in[6];
    P.bc   = (const float*)d_in[7];
    const float* Wih2 = (const float*)d_in[8];
    const float* Whh2 = (const float*)d_in[9];
    P.bih2 = (const float*)d_in[10];
    P.bhh2 = (const float*)d_in[11];
    const float* Wih3 = (const float*)d_in[12];
    const float* Whh3 = (const float*)d_in[13];
    P.bih3 = (const float*)d_in[14];
    P.bhh3 = (const float*)d_in[15];
    const float* Wl = (const float*)d_in[16];
    P.bl   = (const float*)d_in[17];
    P.out  = (float*)d_out;

    // ---- workspace carve (zeroed region first, then weights) ----
    char* ws = (char*)d_ws;
    size_t off = 0;
    auto carve = [&](size_t bytes) -> void* {
        void* p = ws + off;
        off = (off + bytes + 255) & ~(size_t)255;
        return p;
    };
    P.counter = (unsigned*)carve(256);
    P.gen     = P.counter + 1;
    P.kappa = (float*)carve(sizeof(float) * BATCH * KMIX);
    P.h1    = (float*)carve(sizeof(float) * BATCH * H);
    P.h2    = (float*)carve(sizeof(float) * BATCH * H);
    P.h3    = (float*)carve(sizeof(float) * BATCH * H);
    P.gi    = (float*)carve(sizeof(float) * BATCH * GATES);
    P.gh    = (float*)carve(sizeof(float) * BATCH * GATES);
    P.h1b   = (unsigned short*)carve(2u * BATCH * HP);
    P.h2b   = (unsigned short*)carve(2u * BATCH * HP);
    P.h3b   = (unsigned short*)carve(2u * BATCH * HP);
    P.catA2 = (unsigned short*)carve(2u * BATCH * KIN2P);
    P.catA3 = (unsigned short*)carve(2u * BATCH * KIN2P);
    P.catL  = (unsigned short*)carve(2u * BATCH * KOUTP);
    const size_t zeroBytes = off;           // everything above must start at 0
    P.Whh1b = (unsigned short*)carve(2u * GATES * HP);
    P.Whh2b = (unsigned short*)carve(2u * GATES * HP);
    P.Whh3b = (unsigned short*)carve(2u * GATES * HP);
    P.Wih2b = (unsigned short*)carve(2u * GATES * KIN2P);
    P.Wih3b = (unsigned short*)carve(2u * GATES * KIN2P);
    P.Wlb   = (unsigned short*)carve(2u * 128 * KOUTP);

    hipMemsetAsync(d_ws, 0, zeroBytes, stream);

    const int cb = 256, cg = 512;
    convert_pad<<<cg, cb, 0, stream>>>(Whh1, P.Whh1b, GATES, GATES, H, HP);
    convert_pad<<<cg, cb, 0, stream>>>(Whh2, P.Whh2b, GATES, GATES, H, HP);
    convert_pad<<<cg, cb, 0, stream>>>(Whh3, P.Whh3b, GATES, GATES, H, HP);
    convert_pad<<<cg, cb, 0, stream>>>(Wih2, P.Wih2b, GATES, GATES, KIN2, KIN2P);
    convert_pad<<<cg, cb, 0, stream>>>(Wih3, P.Wih3b, GATES, GATES, KIN2, KIN2P);
    convert_pad<<<cg, cb, 0, stream>>>(Wl,   P.Wlb,   128,   OUTD,  KOUT, KOUTP);

    P.nwg = 32;
    rnn_persistent<<<P.nwg, 256, 0, stream>>>(P);
}
